// EncoderRNN_16999480558122
// MI455X (gfx1250) — compile-verified, matrix operand order checked by reading, not run on hardware
//
#include <hip/hip_runtime.h>
#include <hip/hip_bf16.h>
#include <math.h>

// Bidirectional GRU encoder (B=64, T=512, E=H=512) + fused ragged segment-max.
//   1) pack w_ih / w_hh (fp32 -> bf16) into WMMA B-fragment order
//   2) fused embedding-gather + input-projection GEMM (WMMA bf16 -> f32);
//      one block per 16-row m-tile stages the gathered A panel in LDS ONCE and
//      sweeps all 96 n-tiles x 2 directions (24x less gather traffic)
//   3) persistent per-direction/per-m-tile GRU scan: per-step [16,512]x[512,1536]
//      WMMA GEMM, gate math in registers, h carried in registers (fp32) and
//      LDS (bf16), outputs folded directly into segment-max via f32 atomic max.
//   4) zero-fill pass contributes the ctx==0 padded rows inside segments.
// K-loops unrolled x4 so 4 independent load->wmma chains pipeline under the
// split wait counters (partial s_wait_loadcnt_dscnt instead of wait-to-zero).

typedef __bf16 bf16_t;
typedef __attribute__((ext_vector_type(16))) __bf16 v16bf;
typedef __attribute__((ext_vector_type(8)))  float  v8f;

#define Bsz   64
#define Tsz   512
#define Esz   512
#define Hsz   512
#define Gsz   1536        // 3*H
#define NSEG  64
#define KCN   16          // 512 / 32 k-chunks per WMMA K loop
#define NTG   96          // 1536 / 16 n-tiles

union AFrag { v16bf v; unsigned u[8]; };

static __device__ __forceinline__ float sigmoidf_(float x) {
  return 1.0f / (1.0f + __expf(-x));
}

static __device__ __forceinline__ void atomic_max_f32(float* p, float v) {
  // lowers to global_atomic_max_num_f32 (non-returning) on gfx1250
  __hip_atomic_fetch_max(p, v, __ATOMIC_RELAXED, __HIP_MEMORY_SCOPE_AGENT);
}

// ---------------------------------------------------------------------------
// Pack W [1536][512] fp32 row-major into WMMA B-fragment order, bf16:
// P[((nt*KCN + kc)*32 + lane)*16 + jj] = W[n][k] with the ISA 16-bit B layout
// (n = nt*16 + lane&15; k pairs split across lane halves).
// ---------------------------------------------------------------------------
__global__ void pack_w_kernel(const float* __restrict__ W, bf16_t* __restrict__ P) {
  int idx  = blockIdx.x * blockDim.x + threadIdx.x;  // < 96*16*32*16
  int jj   = idx & 15;
  int lane = (idx >> 4) & 31;
  int kc   = (idx >> 9) & 15;
  int nt   = idx >> 13;
  int n    = nt * 16 + (lane & 15);
  int khalf = lane >> 4;
  int v = jj >> 1, lo = jj & 1;
  int k = kc * 32 + ((v >> 2) << 4) + khalf * 8 + ((v & 3) << 1) + lo;
  P[idx] = (bf16_t)W[(size_t)n * 512 + k];
}

__global__ void init_maxed_kernel(float* __restrict__ out) {
  int idx = blockIdx.x * blockDim.x + threadIdx.x;
  if (idx < Bsz * NSEG * 2 * Hsz) out[idx] = -3.402823466e38f;
}

// seg_id[b][t] = segment containing t, or -1 if t outside [0, layout[b][NSEG])
__global__ void segid_kernel(const int* __restrict__ layout, int* __restrict__ segid) {
  int idx = blockIdx.x * blockDim.x + threadIdx.x;   // B*T
  int b = idx >> 9, t = idx & 511;
  const int* L = layout + b * (NSEG + 1);
  int s = -1;
  if (t < L[NSEG]) {
    s = 0;
#pragma unroll 1
    for (int i = 1; i < NSEG; ++i) s += (t >= L[i]) ? 1 : 0;
  }
  segid[idx] = s;
}

// ---------------------------------------------------------------------------
// Fused embedding gather + input projection:
// xp[t][b][n] = emb_table[seq[b][t]] . w_ih[n] + b_ih[n]   (bf16 out)
// One block (256 thr = 8 waves) per 16-row m-tile. The gathered 16x512 bf16
// A panel is staged in LDS once; each wave sweeps 24 of the 192 (n-tile,dir)
// jobs, reading B panels from the L2-resident packed weights.
// ---------------------------------------------------------------------------
__global__ __launch_bounds__(256) void embed_project_kernel(
    const int* __restrict__ seqs, const float* __restrict__ emb,
    const bf16_t* __restrict__ Pf, const bf16_t* __restrict__ Pb,
    const float* __restrict__ bih_f, const float* __restrict__ bih_b,
    bf16_t* __restrict__ xp_f, bf16_t* __restrict__ xp_b) {
  __shared__ bf16_t As[16][520];
  const int tid = threadIdx.x;
  const int m0 = blockIdx.x * 16;

  for (int i = tid; i < 16 * 512; i += 256) {
    int r = i >> 9, k = i & 511;
    int tok = seqs[m0 + r];
    As[r][k] = (bf16_t)emb[(size_t)tok * Esz + k];
  }
  __syncthreads();

  const int wave = tid >> 5, lane = tid & 31;
  const int lm = lane & 15, khalf = lane >> 4;

  // row -> (b, t) decode for the D-layout store (m = b*T + t)
  int bvv[8], tvv[8];
#pragma unroll
  for (int v = 0; v < 8; ++v) {
    int m = m0 + v + ((lane >> 4) << 3);
    bvv[v] = m >> 9;
    tvv[v] = m & 511;
  }

#pragma unroll 1
  for (int jw = wave; jw < 2 * NTG; jw += 8) {
    const int dir = (jw >= NTG) ? 1 : 0;
    const int ntile = dir ? (jw - NTG) : jw;
    const bf16_t* P   = dir ? Pb    : Pf;
    const float*  bih = dir ? bih_b : bih_f;
    bf16_t*       xp  = dir ? xp_b  : xp_f;
    const int n0 = ntile * 16;

    v8f acc;
    {
      float bias = bih[n0 + lm];
#pragma unroll
      for (int i = 0; i < 8; ++i) acc[i] = bias;
    }

    // prefetch the next job's first B panel (global_prefetch_b8)
    {
      int jn = jw + 8;
      if (jn < 2 * NTG) {
        const bf16_t* Pn = (jn >= NTG) ? Pb : Pf;
        int ntn = (jn >= NTG) ? jn - NTG : jn;
        __builtin_prefetch(Pn + ((size_t)ntn * KCN * 32 + lane) * 16, 0, 1);
      }
    }

    // x4 unroll: 4 independent ds/global load -> wmma chains in flight
#pragma unroll 4
    for (int kc = 0; kc < KCN; ++kc) {
      AFrag a;
#pragma unroll
      for (int v = 0; v < 8; ++v) {
        int k0 = kc * 32 + ((v >> 2) << 4) + khalf * 8 + ((v & 3) << 1);
        a.u[v] = *(const unsigned*)&As[lm][k0];
      }
      v16bf bv = *(const v16bf*)(P + (((size_t)ntile * KCN + kc) * 32 + lane) * 16);
      acc = __builtin_amdgcn_wmma_f32_16x16x32_bf16(false, a.v, false, bv,
                                                    (short)0, acc, false, false);
    }

#pragma unroll
    for (int v = 0; v < 8; ++v)
      xp[((size_t)tvv[v] * Bsz + bvv[v]) * Gsz + n0 + lm] = (bf16_t)acc[v];
  }
}

// Padded rows inside a segment contribute ctx == 0 to the max (both directions)
__global__ void zero_fill_kernel(const int* __restrict__ lens,
                                 const int* __restrict__ layout,
                                 const int* __restrict__ segid,
                                 float* __restrict__ maxed) {
  int idx = blockIdx.x * blockDim.x + threadIdx.x;  // B*T*4
  int bt = idx >> 2, chunk = idx & 3;
  int b = bt >> 9, t = bt & 511;
  int len = lens[b];
  int end = layout[b * (NSEG + 1) + NSEG];
  if (t >= len && t < end) {
    int s = segid[bt];
    float* mrow = maxed + ((size_t)b * NSEG + s) * (2 * Hsz) + chunk * 256;
#pragma unroll 1
    for (int c = 0; c < 256; ++c) atomic_max_f32(mrow + c, 0.0f);
  }
}

// ---------------------------------------------------------------------------
// Persistent GRU scan. Grid (4 m-tiles, 2 dirs), 512 thr = 16 waves.
// Each wave owns two n-tiles (nt = 2*wave+{0,1}) and keeps h fp32 in
// registers for its 16x16 patches; each step: hp = h @ w_hh^T + b_hh via
// 3 WMMA accumulators (r/z/n land elementwise-aligned), gate math, masked
// update, bf16 h broadcast through LDS, fused atomic segment-max output.
// Backward direction reads/writes at t_src = len-1-t (matches rev_idx).
// ---------------------------------------------------------------------------
__global__ __launch_bounds__(512) void gru_scan_kernel(
    const bf16_t* __restrict__ xp_f, const bf16_t* __restrict__ xp_b,
    const bf16_t* __restrict__ Whh_f, const bf16_t* __restrict__ Whh_b,
    const float* __restrict__ bhh_f, const float* __restrict__ bhh_b,
    const int* __restrict__ lens, const int* __restrict__ segid,
    float* __restrict__ maxed, float* __restrict__ hidden_out) {
  __shared__ bf16_t Hs[16][520];
  const int tid = threadIdx.x;
  const int dir = blockIdx.y;
  const int b0  = blockIdx.x * 16;
  const bf16_t* xp  = dir ? xp_b  : xp_f;
  const bf16_t* Whh = dir ? Whh_b : Whh_f;
  const float*  bhh = dir ? bhh_b : bhh_f;

  for (int i = tid; i < 16 * 520; i += 512) ((bf16_t*)Hs)[i] = (bf16_t)0.0f;

  const int wave = tid >> 5, lane = tid & 31;
  const int lm = lane & 15, khalf = lane >> 4, mhalf = lane >> 4;

  int brow[8], lenv[8];
#pragma unroll
  for (int v = 0; v < 8; ++v) {
    brow[v] = b0 + v + (mhalf << 3);
    lenv[v] = lens[brow[v]];
  }

  v8f hst[2];
#pragma unroll
  for (int j = 0; j < 2; ++j)
#pragma unroll
    for (int v = 0; v < 8; ++v) hst[j][v] = 0.0f;

  __syncthreads();

  for (int t = 0; t < Tsz; ++t) {
#pragma unroll
    for (int j = 0; j < 2; ++j) {
      const int nt = wave * 2 + j;
      const int n0 = nt * 16;
      const int ncol = n0 + lm;
      v8f acc0, acc1, acc2;
      {
        float br = bhh[ncol], bz = bhh[512 + ncol], bn = bhh[1024 + ncol];
#pragma unroll
        for (int v = 0; v < 8; ++v) { acc0[v] = br; acc1[v] = bz; acc2[v] = bn; }
      }
      // prefetch next step's xp gate row (hides HBM latency in the serial scan)
      if (t + 1 < lenv[0]) {
        int tsn = dir ? (lenv[0] - 2 - t) : (t + 1);
        __builtin_prefetch(xp + ((size_t)tsn * Bsz + brow[0]) * Gsz + ncol, 0, 1);
      }
      // x4 unroll: keeps 8+ b128 loads outstanding across 12 wmma issues
#pragma unroll 4
      for (int kc = 0; kc < KCN; ++kc) {
        AFrag a;
#pragma unroll
        for (int v = 0; v < 8; ++v) {
          int k0 = kc * 32 + ((v >> 2) << 4) + khalf * 8 + ((v & 3) << 1);
          a.u[v] = *(const unsigned*)&Hs[lm][k0];
        }
        v16bf bv0 = *(const v16bf*)(Whh + (((size_t)(nt)      * KCN + kc) * 32 + lane) * 16);
        acc0 = __builtin_amdgcn_wmma_f32_16x16x32_bf16(false, a.v, false, bv0,
                                                       (short)0, acc0, false, false);
        v16bf bv1 = *(const v16bf*)(Whh + (((size_t)(32 + nt) * KCN + kc) * 32 + lane) * 16);
        acc1 = __builtin_amdgcn_wmma_f32_16x16x32_bf16(false, a.v, false, bv1,
                                                       (short)0, acc1, false, false);
        v16bf bv2 = *(const v16bf*)(Whh + (((size_t)(64 + nt) * KCN + kc) * 32 + lane) * 16);
        acc2 = __builtin_amdgcn_wmma_f32_16x16x32_bf16(false, a.v, false, bv2,
                                                       (short)0, acc2, false, false);
      }
#pragma unroll
      for (int v = 0; v < 8; ++v) {
        if (t < lenv[v]) {
          int ts = dir ? (lenv[v] - 1 - t) : t;
          size_t xbase = ((size_t)ts * Bsz + brow[v]) * Gsz;
          float xr = (float)xp[xbase + ncol];
          float xz = (float)xp[xbase + 512 + ncol];
          float xn = (float)xp[xbase + 1024 + ncol];
          float r  = sigmoidf_(xr + acc0[v]);
          float z  = sigmoidf_(xz + acc1[v]);
          float nn = tanhf(xn + r * acc2[v]);
          float hnew = (1.0f - z) * nn + z * hst[j][v];
          hst[j][v] = hnew;
          int sg = segid[brow[v] * Tsz + ts];
          if (sg >= 0)
            atomic_max_f32(maxed + ((size_t)brow[v] * NSEG + sg) * (2 * Hsz)
                                 + dir * Hsz + ncol, hnew);
        }
      }
    }
    __syncthreads();
#pragma unroll
    for (int j = 0; j < 2; ++j) {
      const int n0 = (wave * 2 + j) * 16;
#pragma unroll
      for (int v = 0; v < 8; ++v)
        Hs[v + (mhalf << 3)][n0 + lm] = (bf16_t)hst[j][v];
    }
    __syncthreads();
  }

#pragma unroll
  for (int j = 0; j < 2; ++j) {
    const int n0 = (wave * 2 + j) * 16;
#pragma unroll
    for (int v = 0; v < 8; ++v)
      hidden_out[(size_t)dir * Bsz * Hsz + (size_t)brow[v] * Hsz + n0 + lm] =
          hst[j][v];
  }
}

// ---------------------------------------------------------------------------
extern "C" void kernel_launch(void* const* d_in, const int* in_sizes, int n_in,
                              void* d_out, int out_size, void* d_ws, size_t ws_size,
                              hipStream_t stream) {
  const int*   seqs   = (const int*)d_in[0];
  const int*   lens   = (const int*)d_in[1];
  // d_in[2] hidden: unused by forward
  const int*   layout = (const int*)d_in[3];
  const float* emb    = (const float*)d_in[4];
  const float* w_ih_f = (const float*)d_in[5];
  const float* w_hh_f = (const float*)d_in[6];
  const float* b_ih_f = (const float*)d_in[7];
  const float* b_hh_f = (const float*)d_in[8];
  const float* w_ih_b = (const float*)d_in[9];
  const float* w_hh_b = (const float*)d_in[10];
  const float* b_ih_b = (const float*)d_in[11];
  const float* b_hh_b = (const float*)d_in[12];

  float* maxed      = (float*)d_out;                          // [B,NSEG,2H]
  float* hidden_out = maxed + (size_t)Bsz * NSEG * 2 * Hsz;   // [2,B,H]

  char* ws = (char*)d_ws;
  const size_t PACKN = (size_t)NTG * KCN * 32 * 16;           // 786432 elems
  const size_t PACKB = PACKN * sizeof(bf16_t);                // 1.5 MB
  bf16_t* Pih_f = (bf16_t*)(ws);
  bf16_t* Pih_b = (bf16_t*)(ws + PACKB);
  bf16_t* Phh_f = (bf16_t*)(ws + 2 * PACKB);
  bf16_t* Phh_b = (bf16_t*)(ws + 3 * PACKB);
  const size_t XPB = (size_t)Tsz * Bsz * Gsz * sizeof(bf16_t); // 96 MB
  bf16_t* xp_f = (bf16_t*)(ws + 4 * PACKB);
  bf16_t* xp_b = (bf16_t*)(ws + 4 * PACKB + XPB);
  int*    segid = (int*)(ws + 4 * PACKB + 2 * XPB);

  init_maxed_kernel<<<(Bsz * NSEG * 2 * Hsz) / 256, 256, 0, stream>>>(maxed);
  pack_w_kernel<<<(int)(PACKN / 256), 256, 0, stream>>>(w_ih_f, Pih_f);
  pack_w_kernel<<<(int)(PACKN / 256), 256, 0, stream>>>(w_ih_b, Pih_b);
  pack_w_kernel<<<(int)(PACKN / 256), 256, 0, stream>>>(w_hh_f, Phh_f);
  pack_w_kernel<<<(int)(PACKN / 256), 256, 0, stream>>>(w_hh_b, Phh_b);
  segid_kernel<<<(Bsz * Tsz) / 256, 256, 0, stream>>>(layout, segid);

  // one block per 16-row m-tile; A gathered once, all n-tiles/dirs swept
  embed_project_kernel<<<(Bsz * Tsz) / 16, 256, 0, stream>>>(
      seqs, emb, Pih_f, Pih_b, b_ih_f, b_ih_b, xp_f, xp_b);

  zero_fill_kernel<<<(Bsz * Tsz * 4) / 256, 256, 0, stream>>>(lens, layout, segid, maxed);

  dim3 gScan(4, 2);
  gru_scan_kernel<<<gScan, 512, 0, stream>>>(xp_f, xp_b, Phh_f, Phh_b,
                                             b_hh_f, b_hh_b, lens, segid,
                                             maxed, hidden_out);

  (void)in_sizes; (void)n_in; (void)out_size; (void)ws_size;
}
